// ANI2x_55052890800230
// MI455X (gfx1250) — compile-verified, matrix operand order checked by reading, not run on hardware
//
#include <hip/hip_runtime.h>
#include <math.h>

typedef __bf16 bfloat;
typedef __attribute__((ext_vector_type(16))) __bf16 v16bf;
typedef __attribute__((ext_vector_type(8)))  __bf16 bf16x8;
typedef __attribute__((ext_vector_type(8)))  float  v8f;

#define RCR_  5.1f
#define RCA_  3.5f
#define ETAR_ 19.7f
#define ETAA_ 12.5f
#define ZETA_ 14.1f
#define PI_   3.14159265358979323846f
#define AEVD  1008
#define AEVP  1024
#define NSPC  7

__device__ __forceinline__ float fcutf(float d, float rc) {
    return d < rc ? 0.5f * __cosf(PI_ * d / rc) + 0.5f : 0.0f;
}
__device__ __forceinline__ float dfcutf(float d, float rc) {
    return d < rc ? -0.5f * (PI_ / rc) * __sinf(PI_ * d / rc) : 0.0f;
}

// ---------------------------------------------------------------- utilities
__global__ void zero_f32(float* p, long n) {
    long i = (long)blockIdx.x * blockDim.x + threadIdx.x;
    long st = (long)gridDim.x * blockDim.x;
    for (; i < n; i += st) p[i] = 0.0f;
}

// Deterministic per-species atom lists; padded with atom 0 so tile gathers
// never go out of bounds (stores are still guarded by cnt).
__global__ void build_splist(const int* sgp, int* sp_list, int* sp_cnt, int N) {
    int s = threadIdx.x;
    if (s >= NSPC) return;
    int c = 0;
    for (int a = 0; a < N; a++)
        if (sgp[a] == s) sp_list[s * N + c++] = a;
    sp_cnt[s] = c;
    for (; c < N; c++) sp_list[s * N + c] = 0;
}

// ---------------------------------------------------- weight packing (bf16)
// Transpose+cast: out[n*ldo + k] = (k < R) ? W[k*C + n] : 0   (n<C, k<ldo)
__global__ void pack_wt(const float* __restrict__ W, bfloat* __restrict__ out,
                        int R, int C, int ldo) {
    int t = blockIdx.x * blockDim.x + threadIdx.x;
    if (t >= C * ldo) return;
    int n = t / ldo, k = t - n * ldo;
    out[t] = (bfloat)(k < R ? W[(size_t)k * C + n] : 0.0f);
}
// Plain cast copy (row-major preserved)
__global__ void pack_wb(const float* __restrict__ W, bfloat* __restrict__ out,
                        int total) {
    int t = blockIdx.x * blockDim.x + threadIdx.x;
    if (t < total) out[t] = (bfloat)W[t];
}

// ------------------------------------------------------- neighbor list (RCA)
__global__ void nbr_build(const float* dist, const int* ai, const int* aj,
                          int* ncnt, int* slot, int P) {
    int hp = blockIdx.x * blockDim.x + threadIdx.x;
    if (hp >= 2 * P) return;
    int p = hp < P ? hp : hp - P;
    float d = dist[p];
    if (d >= RCA_) return;
    int ci = hp < P ? ai[p] : aj[p];
    int r = atomicAdd(&ncnt[ci], 1);
    if (r < 32) slot[ci * 32 + r] = hp;
}

__global__ void nbr_sort(int* slot, const int* ncnt, int N) {
    int a = blockIdx.x * blockDim.x + threadIdx.x;
    if (a >= N) return;
    int k = ncnt[a]; if (k > 32) k = 32;
    int* s = slot + a * 32;
    for (int i = 1; i < k; i++) {
        int v = s[i]; int j = i - 1;
        while (j >= 0 && s[j] > v) { s[j + 1] = s[j]; j--; }
        s[j + 1] = v;
    }
}

// ------------------------------------------------------------- radial AEV
__global__ void radial_fwd(const float* dist, const int* ai, const int* aj,
                           const int* sp, float* aev, int P) {
    int hp = blockIdx.x * blockDim.x + threadIdx.x;
    if (hp >= 2 * P) return;
    int p = hp < P ? hp : hp - P;
    float d = dist[p];
    if (d >= RCR_) return;
    int ci = hp < P ? ai[p] : aj[p];
    int nb = hp < P ? aj[p] : ai[p];
    float fc = fcutf(d, RCR_);
    float* base = &aev[(size_t)ci * AEVD + sp[nb] * 16];
#pragma unroll
    for (int t = 0; t < 16; t++) {
        float sh = 0.8f + 0.26875f * t;
        float e = __expf(-ETAR_ * (d - sh) * (d - sh));
        atomicAdd(&base[t], 0.25f * e * fc);
    }
}

__global__ void radial_bwd(const float* diff, const float* dist, const int* ai,
                           const int* aj, const int* sp, const float* daev,
                           float* grad, int P) {
    int hp = blockIdx.x * blockDim.x + threadIdx.x;
    if (hp >= 2 * P) return;
    int p = hp < P ? hp : hp - P;
    float d = dist[p];
    if (d >= RCR_ || d <= 0.0f) return;
    int ci = hp < P ? ai[p] : aj[p];
    int nb = hp < P ? aj[p] : ai[p];
    float fc = fcutf(d, RCR_);
    float dfc = dfcutf(d, RCR_);
    const float* G = &daev[(size_t)ci * AEVD + sp[nb] * 16];
    float gd = 0.0f;
#pragma unroll
    for (int t = 0; t < 16; t++) {
        float sh = 0.8f + 0.26875f * t;
        float dm = d - sh;
        float e = __expf(-ETAR_ * dm * dm);
        gd += G[t] * 0.25f * (-2.0f * ETAR_ * dm * e * fc + e * dfc);
    }
    float sg = hp < P ? 1.0f : -1.0f;
    float inv = gd * sg / d;
    float gx = inv * diff[3 * p + 0];
    float gy = inv * diff[3 * p + 1];
    float gz = inv * diff[3 * p + 2];
    atomicAdd(&grad[nb * 3 + 0],  gx); atomicAdd(&grad[nb * 3 + 1],  gy);
    atomicAdd(&grad[nb * 3 + 2],  gz);
    atomicAdd(&grad[ci * 3 + 0], -gx); atomicAdd(&grad[ci * 3 + 1], -gy);
    atomicAdd(&grad[ci * 3 + 2], -gz);
}

// ------------------------------------------------------------- angular AEV
__global__ void angular_fwd(const float* diff, const float* dist, const int* ai,
                            const int* aj, const int* sp, const int* slot,
                            const int* ncnt, float* aev, int P) {
    int atom = blockIdx.x;
    int tid = threadIdx.x;
    __shared__ float sV[32][3]; __shared__ float sD[32]; __shared__ int sS[32];
    __shared__ float acc[896];
    for (int t = tid; t < 896; t += blockDim.x) acc[t] = 0.0f;
    int cnt = ncnt[atom]; if (cnt > 32) cnt = 32;
    if (tid < cnt) {
        int hp = slot[atom * 32 + tid];
        int p = hp < P ? hp : hp - P;
        float sg = hp < P ? 1.0f : -1.0f;
        sV[tid][0] = sg * diff[3 * p + 0];
        sV[tid][1] = sg * diff[3 * p + 1];
        sV[tid][2] = sg * diff[3 * p + 2];
        sD[tid] = dist[p];
        int nb = hp < P ? aj[p] : ai[p];
        sS[tid] = sp[nb];
    }
    __syncthreads();
    int npairs = cnt * (cnt - 1) / 2;
    for (int q = tid; q < npairs; q += blockDim.x) {
        int m = 0, qq = q;
        while (qq >= cnt - 1 - m) { qq -= cnt - 1 - m; m++; }
        int n = m + 1 + qq;
        float Dm = sD[m], Dn = sD[n];
        float dot = sV[m][0] * sV[n][0] + sV[m][1] * sV[n][1] + sV[m][2] * sV[n][2];
        float dd = fmaxf(Dm * Dn, 1e-10f);
        float c = dot / dd;
        float ang = acosf(0.95f * c);
        float w = 2.0f * fcutf(Dm, RCA_) * fcutf(Dn, RCA_);
        float davg = 0.5f * (Dm + Dn);
        int smin = sS[m] < sS[n] ? sS[m] : sS[n];
        int smax = sS[m] < sS[n] ? sS[n] : sS[m];
        int pbin = smin * NSPC + smax - (smin * (smin + 1)) / 2;
        float f1v[8], f2v[4];
#pragma unroll
        for (int z = 0; z < 8; z++) {
            float u = ang - (z + 0.5f) * (PI_ / 8.0f);
            float p5 = 0.5f * (1.0f + __cosf(u));
            f1v[z] = __powf(p5, ZETA_);
        }
#pragma unroll
        for (int a2 = 0; a2 < 4; a2++) {
            float t = davg - (0.8f + 0.675f * a2);
            f2v[a2] = __expf(-ETAA_ * t * t);
        }
        float* accb = &acc[pbin * 32];
#pragma unroll
        for (int z = 0; z < 8; z++)
#pragma unroll
            for (int a2 = 0; a2 < 4; a2++)
                atomicAdd(&accb[z * 4 + a2], w * f1v[z] * f2v[a2]);
    }
    __syncthreads();
    for (int t = tid; t < 896; t += blockDim.x)
        aev[(size_t)atom * AEVD + 112 + t] = acc[t];
}

__global__ void angular_bwd(const float* diff, const float* dist, const int* ai,
                            const int* aj, const int* sp, const int* slot,
                            const int* ncnt, const float* daev, float* grad, int P) {
    int atom = blockIdx.x;
    int tid = threadIdx.x;
    __shared__ float sV[32][3]; __shared__ float sD[32];
    __shared__ int sS[32]; __shared__ int sA[32];
    __shared__ float gsl[896];
    for (int t = tid; t < 896; t += blockDim.x)
        gsl[t] = daev[(size_t)atom * AEVD + 112 + t];
    int cnt = ncnt[atom]; if (cnt > 32) cnt = 32;
    if (tid < cnt) {
        int hp = slot[atom * 32 + tid];
        int p = hp < P ? hp : hp - P;
        float sg = hp < P ? 1.0f : -1.0f;
        sV[tid][0] = sg * diff[3 * p + 0];
        sV[tid][1] = sg * diff[3 * p + 1];
        sV[tid][2] = sg * diff[3 * p + 2];
        sD[tid] = dist[p];
        int nb = hp < P ? aj[p] : ai[p];
        sS[tid] = sp[nb]; sA[tid] = nb;
    }
    __syncthreads();
    int npairs = cnt * (cnt - 1) / 2;
    for (int q = tid; q < npairs; q += blockDim.x) {
        int m = 0, qq = q;
        while (qq >= cnt - 1 - m) { qq -= cnt - 1 - m; m++; }
        int n = m + 1 + qq;
        float Dm = sD[m], Dn = sD[n];
        float vmx = sV[m][0], vmy = sV[m][1], vmz = sV[m][2];
        float vnx = sV[n][0], vny = sV[n][1], vnz = sV[n][2];
        float dot = vmx * vnx + vmy * vny + vmz * vnz;
        float dd = fmaxf(Dm * Dn, 1e-10f);
        float c = dot / dd;
        float cc = 0.95f * c;
        float ang = acosf(cc);
        float fm = fcutf(Dm, RCA_), fn = fcutf(Dn, RCA_);
        float w = 2.0f * fm * fn;
        float davg = 0.5f * (Dm + Dn);
        int smin = sS[m] < sS[n] ? sS[m] : sS[n];
        int smax = sS[m] < sS[n] ? sS[n] : sS[m];
        int pbin = smin * NSPC + smax - (smin * (smin + 1)) / 2;
        const float* G = &gsl[pbin * 32];
        float f1v[8], df1v[8], f2v[4], dt2v[4];
#pragma unroll
        for (int z = 0; z < 8; z++) {
            float u = ang - (z + 0.5f) * (PI_ / 8.0f);
            float p5 = 0.5f * (1.0f + __cosf(u));
            f1v[z] = __powf(p5, ZETA_);
            df1v[z] = -ZETA_ * __powf(p5, ZETA_ - 1.0f) * 0.5f * __sinf(u);
        }
#pragma unroll
        for (int a2 = 0; a2 < 4; a2++) {
            float t = davg - (0.8f + 0.675f * a2);
            f2v[a2] = __expf(-ETAA_ * t * t);
            dt2v[a2] = -2.0f * ETAA_ * t * f2v[a2];
        }
        float Sw = 0.0f, Sang = 0.0f, Sdavg = 0.0f;
#pragma unroll
        for (int z = 0; z < 8; z++) {
            float Az = 0.0f;
#pragma unroll
            for (int a2 = 0; a2 < 4; a2++) Az += G[z * 4 + a2] * f2v[a2];
            Sw += Az * f1v[z];
            Sang += Az * df1v[z];
        }
#pragma unroll
        for (int a2 = 0; a2 < 4; a2++) {
            float Ba = 0.0f;
#pragma unroll
            for (int z = 0; z < 8; z++) Ba += G[z * 4 + a2] * f1v[z];
            Sdavg += Ba * dt2v[a2];
        }
        Sang *= w; Sdavg *= w;
        float dLdc = Sang * (-0.95f * rsqrtf(1.0f - cc * cc));
        float dLdDm = Sw * 2.0f * dfcutf(Dm, RCA_) * fn + 0.5f * Sdavg - dLdc * c / Dm;
        float dLdDn = Sw * 2.0f * fm * dfcutf(Dn, RCA_) + 0.5f * Sdavg - dLdc * c / Dn;
        float cv = dLdc / dd;
        float km = dLdDm / Dm, kn = dLdDn / Dn;
        float gmx = cv * vnx + km * vmx, gmy = cv * vny + km * vmy, gmz = cv * vnz + km * vmz;
        float gnx = cv * vmx + kn * vnx, gny = cv * vmy + kn * vny, gnz = cv * vmz + kn * vnz;
        atomicAdd(&grad[sA[m] * 3 + 0], gmx); atomicAdd(&grad[sA[m] * 3 + 1], gmy);
        atomicAdd(&grad[sA[m] * 3 + 2], gmz);
        atomicAdd(&grad[sA[n] * 3 + 0], gnx); atomicAdd(&grad[sA[n] * 3 + 1], gny);
        atomicAdd(&grad[sA[n] * 3 + 2], gnz);
        atomicAdd(&grad[atom * 3 + 0], -(gmx + gnx));
        atomicAdd(&grad[atom * 3 + 1], -(gmy + gny));
        atomicAdd(&grad[atom * 3 + 2], -(gmz + gnz));
    }
}

// ------------------------------------------------- AEV f32 -> bf16 (padded)
__global__ void aev_to_bf16(const float* aev, bfloat* out, int N) {
    int t = blockIdx.x * blockDim.x + threadIdx.x;
    if (t >= N * AEVP) return;
    int a = t >> 10, col = t & (AEVP - 1);
    out[t] = (bfloat)(col < AEVD ? aev[(size_t)a * AEVD + col] : 0.0f);
}

// ------------------------------------------------------------ WMMA GEMM core
// Fragment (A or B): lane = row (0..15 mirrored), holds K = klo..klo+7 and
// klo+16..klo+23; two 16-byte loads.
__device__ __forceinline__ v16bf load_frag(const bfloat* __restrict__ M, int ld,
                                           int row, int klo) {
    const bfloat* p = M + (size_t)row * ld + klo;
    bf16x8 lo = *(const bf16x8*)p;
    bf16x8 hi = *(const bf16x8*)(p + 16);
    v16bf f;
#pragma unroll
    for (int i = 0; i < 8; i++) { f[i] = lo[i]; f[8 + i] = hi[i]; }
    return f;
}

// One wave per 16x(NT*16) output tile; A fragment reused NT times per K step.
// Both operands are pre-packed bf16 with K contiguous per row (B = W^T for
// forward, B = W for backward), so the loop is branch- and convert-free.
// EMODE 0: bias + CELU(0.1) -> bf16 activations
// EMODE 1: multiply by celu'(href) -> bf16 delta
// EMODE 2: accumulate into daev[atom] (row gathered via rows_map)
template <int NT, int EMODE>
__global__ __launch_bounds__(32) void mlp_gemm(
    const bfloat* __restrict__ A, int lda,
    const bfloat* __restrict__ B, int ldb,
    int K, int Nout, int ntbase,
    const int* __restrict__ rows_map, const int* __restrict__ cntp,
    int a_gather,
    const float* __restrict__ bias, const bfloat* __restrict__ href,
    bfloat* __restrict__ outb, float* __restrict__ daev) {
    int cnt = *cntp;
    int mt = blockIdx.x;
    if (mt * 16 >= cnt) return;
    int nt0 = ntbase + blockIdx.y * (NT * 16);
    int lane = threadIdx.x;
    int half = lane >> 4;
    int lrow = mt * 16 + (lane & 15);
    int arow = a_gather ? rows_map[lrow] : lrow;   // list padded; always in-bounds
    int nl = lane & 15;
    v8f acc[NT];
#pragma unroll
    for (int j = 0; j < NT; j++)
        acc[j] = (v8f){0.f, 0.f, 0.f, 0.f, 0.f, 0.f, 0.f, 0.f};

    for (int kt = 0; kt < K; kt += 32) {
        int klo = kt + half * 8;
        v16bf af = load_frag(A, lda, arow, klo);
#pragma unroll
        for (int j = 0; j < NT; j++) {
            v16bf bf = load_frag(B, ldb, nt0 + j * 16 + nl, klo);
            acc[j] = __builtin_amdgcn_wmma_f32_16x16x32_bf16(
                false, af, false, bf, (short)0, acc[j], false, false);
        }
    }

#pragma unroll
    for (int j = 0; j < NT; j++) {
        int n = nt0 + j * 16 + nl;
#pragma unroll
        for (int r = 0; r < 8; r++) {
            int drow = mt * 16 + half * 8 + r;  // D: lane = N, vgpr r = M (+8 hi half)
            if (drow >= cnt) continue;
            float v = acc[j][r];
            if (EMODE == 0) {
                v += bias[n];
                float y = v > 0.0f ? v : 0.1f * (__expf(10.0f * v) - 1.0f);
                outb[(size_t)drow * Nout + n] = (bfloat)y;
            } else if (EMODE == 1) {
                float y = (float)href[(size_t)drow * Nout + n];
                float dr = y > 0.0f ? 1.0f : (10.0f * y + 1.0f);
                outb[(size_t)drow * Nout + n] = (bfloat)(v * dr);
            } else {
                int atom = rows_map[drow];
                daev[(size_t)atom * AEVD + n] += v;
            }
        }
    }
}

template <int EMODE>
static void launch_gemm(int mtiles, const bfloat* A, int lda, const bfloat* B,
                        int ldb, int K, int Nout, const int* rmap,
                        const int* cp, int a_gather, const float* bias,
                        const bfloat* href, bfloat* outb, float* daev,
                        hipStream_t stream) {
    int full = Nout / 64;
    int rem = (Nout % 64) / 16;
    if (full)
        mlp_gemm<4, EMODE><<<dim3(mtiles, full), 32, 0, stream>>>(
            A, lda, B, ldb, K, Nout, 0, rmap, cp, a_gather, bias, href, outb, daev);
    int ntb = full * 64;
    if (rem == 1)
        mlp_gemm<1, EMODE><<<dim3(mtiles, 1), 32, 0, stream>>>(
            A, lda, B, ldb, K, Nout, ntb, rmap, cp, a_gather, bias, href, outb, daev);
    else if (rem == 2)
        mlp_gemm<2, EMODE><<<dim3(mtiles, 1), 32, 0, stream>>>(
            A, lda, B, ldb, K, Nout, ntb, rmap, cp, a_gather, bias, href, outb, daev);
    else if (rem == 3)
        mlp_gemm<3, EMODE><<<dim3(mtiles, 1), 32, 0, stream>>>(
            A, lda, B, ldb, K, Nout, ntb, rmap, cp, a_gather, bias, href, outb, daev);
}

// ------------------------------------------------------------ output layer
__global__ void out_layer(const bfloat* h3, int H3, const float* W3,
                          const float* b3, const int* rows_map, const int* cntp,
                          float* e_atom) {
    int cnt = *cntp;
    int row = blockIdx.x;
    if (row >= cnt) return;
    int lane = threadIdx.x;
    float s = 0.0f;
    for (int k = lane; k < H3; k += 32) s += (float)h3[(size_t)row * H3 + k] * W3[k];
#pragma unroll
    for (int o = 16; o; o >>= 1) s += __shfl_xor(s, o, 32);
    if (lane == 0) e_atom[rows_map[row]] += s + b3[0];
}

__global__ void seed_g3(const bfloat* h3, int H3, const float* W3,
                        const int* cntp, bfloat* g3) {
    int cnt = *cntp;
    int row = blockIdx.x;
    if (row >= cnt) return;
    for (int k = threadIdx.x; k < H3; k += blockDim.x) {
        float y = (float)h3[(size_t)row * H3 + k];
        float dr = y > 0.0f ? 1.0f : (10.0f * y + 1.0f);
        g3[(size_t)row * H3 + k] = (bfloat)(W3[k] * 0.125f * dr);  // 1/NMODELS
    }
}

// ---------------------------------------------------------------- epilogue
__global__ void write_forces(const float* grad, float* out, int n3) {
    int t = blockIdx.x * blockDim.x + threadIdx.x;
    if (t < n3) out[1 + t] = -grad[t];
}

__global__ void energy_reduce(const float* e_atom, const int* sgp,
                              const float* sae, float* out, int N) {
    __shared__ float red[256];
    float s = 0.0f;
    for (int a = threadIdx.x; a < N; a += 256) {
        int sg = sgp[a];
        float sh = (sg >= 0) ? sae[sg < 6 ? sg : 6] : 0.0f;
        s += e_atom[a] * 0.125f + sh;
    }
    red[threadIdx.x] = s;
    __syncthreads();
    for (int o = 128; o; o >>= 1) {
        if (threadIdx.x < o) red[threadIdx.x] += red[threadIdx.x + o];
        __syncthreads();
    }
    if (threadIdx.x == 0) out[0] = red[0];
}

// ================================================================== driver
extern "C" void kernel_launch(void* const* d_in, const int* in_sizes, int n_in,
                              void* d_out, int out_size, void* d_ws, size_t ws_size,
                              hipStream_t stream) {
    if (n_in < 455) return;  // 6 tensors + 448 params + sae
    const int* species = (const int*)d_in[0];
    const int* ai = (const int*)d_in[2];
    const int N = in_sizes[0];
    const int P = in_sizes[2] / 2;
    const int* aj = ai + P;
    const float* diff = (const float*)d_in[3];
    const float* dist = (const float*)d_in[4];
    const int* sgp = (const int*)d_in[5];
    const float* sae = (const float*)d_in[n_in - 1];
    auto PW = [&](int m, int s, int l) {
        return (const float*)d_in[6 + (((m * 7 + s) * 4) + l) * 2];
    };
    auto PB = [&](int m, int s, int l) {
        return (const float*)d_in[6 + (((m * 7 + s) * 4) + l) * 2 + 1];
    };
    static const int HID[7][3] = {{256,192,160},{224,192,160},{192,160,128},
                                  {192,160,128},{160,128,96},{160,128,96},{160,128,96}};

    // ---- workspace layout
    char* ws = (char*)d_ws;
    size_t off = 0;
    auto walloc = [&](size_t bytes) -> size_t {
        size_t r = off; off = (off + bytes + 255) & ~(size_t)255; return r;
    };
    size_t o_aev  = walloc((size_t)N * AEVD * 4);
    size_t o_daev = walloc((size_t)N * AEVD * 4);
    size_t o_eat  = walloc((size_t)N * 4);
    size_t o_grad = walloc((size_t)N * 3 * 4);
    size_t o_ncnt = walloc((size_t)N * 4);
    size_t o_scnt = walloc(64);
    size_t zero_bytes = off;
    size_t o_baev = walloc((size_t)N * AEVP * 2);
    size_t o_h1   = walloc((size_t)N * 256 * 2);
    size_t o_h2   = walloc((size_t)N * 192 * 2);
    size_t o_h3   = walloc((size_t)N * 160 * 2);
    size_t o_gA   = walloc((size_t)N * 256 * 2);
    size_t o_gB   = walloc((size_t)N * 256 * 2);
    size_t o_slot = walloc((size_t)N * 32 * 4);
    size_t o_slst = walloc((size_t)7 * N * 4);
    size_t o_wt0  = walloc((size_t)256 * AEVP * 2);   // [H1][1024]
    size_t o_wt1  = walloc((size_t)192 * 256 * 2);    // [H2][H1]
    size_t o_wt2  = walloc((size_t)160 * 192 * 2);    // [H3][H2]
    size_t o_wb0  = walloc((size_t)AEVD * 256 * 2);   // [1008][H1]
    size_t o_wb1  = walloc((size_t)256 * 192 * 2);    // [H1][H2]
    size_t o_wb2  = walloc((size_t)192 * 160 * 2);    // [H2][H3]
    if (ws_size < off) return;

    float*  aev   = (float*)(ws + o_aev);
    float*  daev  = (float*)(ws + o_daev);
    float*  eat   = (float*)(ws + o_eat);
    float*  grad  = (float*)(ws + o_grad);
    int*    ncnt  = (int*)(ws + o_ncnt);
    int*    scnt  = (int*)(ws + o_scnt);
    bfloat* baev  = (bfloat*)(ws + o_baev);
    bfloat* h1    = (bfloat*)(ws + o_h1);
    bfloat* h2    = (bfloat*)(ws + o_h2);
    bfloat* h3    = (bfloat*)(ws + o_h3);
    bfloat* gA    = (bfloat*)(ws + o_gA);
    bfloat* gB    = (bfloat*)(ws + o_gB);
    int*    slot  = (int*)(ws + o_slot);
    int*    slist = (int*)(ws + o_slst);
    bfloat* wt0   = (bfloat*)(ws + o_wt0);
    bfloat* wt1   = (bfloat*)(ws + o_wt1);
    bfloat* wt2   = (bfloat*)(ws + o_wt2);
    bfloat* wb0   = (bfloat*)(ws + o_wb0);
    bfloat* wb1   = (bfloat*)(ws + o_wb1);
    bfloat* wb2   = (bfloat*)(ws + o_wb2);

    zero_f32<<<2048, 256, 0, stream>>>((float*)ws, (long)(zero_bytes / 4));
    build_splist<<<1, 8, 0, stream>>>(sgp, slist, scnt, N);
    nbr_build<<<(2 * P + 255) / 256, 256, 0, stream>>>(dist, ai, aj, ncnt, slot, P);
    nbr_sort<<<(N + 127) / 128, 128, 0, stream>>>(slot, ncnt, N);
    radial_fwd<<<(2 * P + 127) / 128, 128, 0, stream>>>(dist, ai, aj, species, aev, P);
    angular_fwd<<<N, 128, 0, stream>>>(diff, dist, ai, aj, species, slot, ncnt, aev, P);
    aev_to_bf16<<<(N * AEVP + 255) / 256, 256, 0, stream>>>(aev, baev, N);

    int mtiles = N / 16;
    for (int m = 0; m < 8; m++) {
        for (int s = 0; s < 7; s++) {
            int H1 = HID[s][0], H2 = HID[s][1], H3d = HID[s][2];
            const int* rmap = slist + s * N;
            const int* cp = scnt + s;
            // pack this pass's weights: W^T (fwd) and W (bwd), both bf16
            pack_wt<<<(H1 * AEVP + 255) / 256, 256, 0, stream>>>(
                PW(m, s, 0), wt0, AEVD, H1, AEVP);
            pack_wt<<<(H2 * H1 + 255) / 256, 256, 0, stream>>>(
                PW(m, s, 1), wt1, H1, H2, H1);
            pack_wt<<<(H3d * H2 + 255) / 256, 256, 0, stream>>>(
                PW(m, s, 2), wt2, H2, H3d, H2);
            pack_wb<<<(AEVD * H1 + 255) / 256, 256, 0, stream>>>(
                PW(m, s, 0), wb0, AEVD * H1);
            pack_wb<<<(H1 * H2 + 255) / 256, 256, 0, stream>>>(
                PW(m, s, 1), wb1, H1 * H2);
            pack_wb<<<(H2 * H3d + 255) / 256, 256, 0, stream>>>(
                PW(m, s, 2), wb2, H2 * H3d);
            // forward
            launch_gemm<0>(mtiles, baev, AEVP, wt0, AEVP, AEVP, H1, rmap, cp, 1,
                           PB(m, s, 0), nullptr, h1, nullptr, stream);
            launch_gemm<0>(mtiles, h1, H1, wt1, H1, H1, H2, rmap, cp, 0,
                           PB(m, s, 1), nullptr, h2, nullptr, stream);
            launch_gemm<0>(mtiles, h2, H2, wt2, H2, H2, H3d, rmap, cp, 0,
                           PB(m, s, 2), nullptr, h3, nullptr, stream);
            out_layer<<<N, 32, 0, stream>>>(h3, H3d, PW(m, s, 3), PB(m, s, 3),
                                            rmap, cp, eat);
            // backward (dgrad only)
            seed_g3<<<N, 128, 0, stream>>>(h3, H3d, PW(m, s, 3), cp, gA);
            launch_gemm<1>(mtiles, gA, H3d, wb2, H3d, H3d, H2, rmap, cp, 0,
                           nullptr, h2, gB, nullptr, stream);
            launch_gemm<1>(mtiles, gB, H2, wb1, H2, H2, H1, rmap, cp, 0,
                           nullptr, h1, gA, nullptr, stream);
            launch_gemm<2>(mtiles, gA, H1, wb0, H1, H1, AEVD, rmap, cp, 0,
                           nullptr, nullptr, nullptr, daev, stream);
        }
    }

    radial_bwd<<<(2 * P + 127) / 128, 128, 0, stream>>>(diff, dist, ai, aj, species,
                                                        daev, grad, P);
    angular_bwd<<<N, 128, 0, stream>>>(diff, dist, ai, aj, species, slot, ncnt,
                                       daev, grad, P);
    write_forces<<<(3 * N + 255) / 256, 256, 0, stream>>>(grad, (float*)d_out, 3 * N);
    energy_reduce<<<1, 256, 0, stream>>>(eat, sgp, sae, (float*)d_out, N);
}
